// MultiheadAttention_76836964926097
// MI455X (gfx1250) — matrix operand order repaired and checked by reading, not compiled, not run on hardware
//
#include <hip/hip_runtime.h>
#include <hip/hip_bf16.h>
#include <math.h>
#include <stdint.h>

#define D_MODEL   2048
#define NUM_HEADS 16
#define KV_HEADS  4
#define HEAD_DIM  128
#define SEQ       4096
#define KV_OUT    512
#define QKV_N     3072   // 2048 (Q) + 512 (K) + 512 (V)

typedef __bf16 bf16;
typedef __attribute__((ext_vector_type(16))) __bf16 bf16x16;
typedef __attribute__((ext_vector_type(8)))  float  f32x8;
typedef __attribute__((ext_vector_type(4)))  float  f32x4v;

union FragU { bf16x16 v; f32x4v f[2]; };

__device__ __forceinline__ f32x8 zero8() {
    f32x8 z;
#pragma unroll
    for (int i = 0; i < 8; ++i) z[i] = 0.0f;
    return z;
}

// ---- CDNA5 async global->LDS copy (16B per lane, tracked by ASYNCcnt) ----
__device__ __forceinline__ void async_copy_b128(void* lds_dst, const void* gsrc) {
    uint32_t l = (uint32_t)(uintptr_t)lds_dst;     // low 32 bits = LDS offset
    uint64_t g = (uint64_t)(uintptr_t)gsrc;
    asm volatile("global_load_async_to_lds_b128 %0, %1, off"
                 :: "v"(l), "v"(g) : "memory");
}
__device__ __forceinline__ void wait_async0() {
    asm volatile("s_wait_asynccnt 0" ::: "memory");
}

// A fragment (16x32 bf16), ISA layout: lanes 0-15 rows m, elems 0..7 = K0..7,
// elems 8..15 = K16..23 ; lanes 16-31 same rows, K8..15 / K24..31.
__device__ __forceinline__ bf16x16 ldsA_frag(const bf16* base, int stride, int lane) {
    int m  = lane & 15;
    int kb = (lane & 16) >> 1;            // 0 or 8
    const bf16* p = base + m * stride + kb;
    FragU u;
    u.f[0] = *(const f32x4v*)(p);         // K kb .. kb+7
    u.f[1] = *(const f32x4v*)(p + 16);    // K kb+16 .. kb+23
    return u.v;
}

// B fragment (32x16 bf16) from LDS stored column-major Bs[n][k]:
// lanes 0-15 col n hold K0..7 / K16..23 ; lanes 16-31 col n hold K8..15 / K24..31.
__device__ __forceinline__ bf16x16 ldsB_frag(const bf16* base, int stride, int lane) {
    int n  = lane & 15;
    int kb = (lane & 16) >> 1;            // 0 or 8
    const bf16* p = base + n * stride + kb;
    FragU u;
    u.f[0] = *(const f32x4v*)(p);         // K kb .. kb+7
    u.f[1] = *(const f32x4v*)(p + 16);    // K kb+16 .. kb+23
    return u.v;
}

__device__ __forceinline__ f32x8 wmma_bf16(bf16x16 a, bf16x16 b, f32x8 c) {
    return __builtin_amdgcn_wmma_f32_16x16x32_bf16(false, a, false, b, (short)0, c,
                                                   false, false);
}

// ---------------------------------------------------------------- converts
__global__ void k_cast_bf16(const float* __restrict__ src, bf16* __restrict__ dst, int n) {
    int i = blockIdx.x * blockDim.x + threadIdx.x;
    if (i < n) dst[i] = (bf16)src[i];
}

// out[n][k] = (bf16) in[k][n]  -- LDS-tiled 32x32 transpose, coalesced both sides
__global__ __launch_bounds__(256)
void k_cast_transpose(const float* __restrict__ in, bf16* __restrict__ out, int K, int N) {
    __shared__ bf16 tile[32][33];
    int kb = blockIdx.y * 32, nb = blockIdx.x * 32;
    int tx = threadIdx.x & 31, ty = threadIdx.x >> 5;   // ty in 0..7
#pragma unroll
    for (int j = ty; j < 32; j += 8)
        tile[j][tx] = (bf16)in[(size_t)(kb + j) * N + nb + tx];
    __syncthreads();
#pragma unroll
    for (int j = ty; j < 32; j += 8)
        out[(size_t)(nb + j) * K + kb + tx] = tile[tx][j];
}

// ---------------------------------------------------------------- GEMM (bf16 WMMA)
// C[M][N] (f32) = A[M][K] (bf16 row-major) x Bt[N][K] (bf16 N-major, pre-transposed)
// block tile 128x128, K-step 32, 256 threads = 8 waves, wave tile 32x64.
// Double-buffered LDS, all tile traffic via async global->LDS DMA.
#define GT_M 128
#define GT_N 128
#define GT_K 32
#define LDT  40   // padded LDS row stride (elements); 80B rows keep 16B align

__global__ __launch_bounds__(256)
void k_gemm_bf16(const bf16* __restrict__ A, const bf16* __restrict__ Bt,
                 float* __restrict__ C, const float* __restrict__ bias,
                 int M, int N, int K) {
    __shared__ bf16 As[2][GT_M * LDT];
    __shared__ bf16 Bs[2][GT_N * LDT];

    int tid = threadIdx.x;
    int wid = tid >> 5, lane = tid & 31;
    int bm = blockIdx.y * GT_M, bn = blockIdx.x * GT_N;
    int wm = (wid >> 1) * 32;     // wave row offset (0..96)
    int wn = (wid & 1) * 64;      // wave col offset (0 or 64)

    auto tileA = [&](int k0, int b) {
#pragma unroll
        for (int i = tid; i < GT_M * 4; i += 256) {
            int r = i >> 2, c = (i & 3) << 3;
            async_copy_b128(&As[b][r * LDT + c], &A[(size_t)(bm + r) * K + k0 + c]);
        }
    };
    auto tileB = [&](int k0, int b) {
#pragma unroll
        for (int i = tid; i < GT_N * 4; i += 256) {
            int r = i >> 2, c = (i & 3) << 3;
            async_copy_b128(&Bs[b][r * LDT + c], &Bt[(size_t)(bn + r) * K + k0 + c]);
        }
    };

    f32x8 acc[2][4];
#pragma unroll
    for (int mi = 0; mi < 2; ++mi)
#pragma unroll
        for (int ni = 0; ni < 4; ++ni) acc[mi][ni] = zero8();

    // prologue: stage tile 0
    tileA(0, 0);
    tileB(0, 0);
    wait_async0();
    __syncthreads();

    int buf = 0;
    for (int k0 = 0; k0 < K; k0 += GT_K) {
        int kn = k0 + GT_K;
        if (kn < K) {               // prefetch next tile while computing
            tileA(kn, buf ^ 1);
            tileB(kn, buf ^ 1);
        }

        bf16x16 af[2], bfv[4];
#pragma unroll
        for (int mi = 0; mi < 2; ++mi)
            af[mi] = ldsA_frag(&As[buf][(wm + 16 * mi) * LDT], LDT, lane);
#pragma unroll
        for (int ni = 0; ni < 4; ++ni)
            bfv[ni] = ldsB_frag(&Bs[buf][(wn + 16 * ni) * LDT], LDT, lane);
#pragma unroll
        for (int mi = 0; mi < 2; ++mi)
#pragma unroll
            for (int ni = 0; ni < 4; ++ni)
                acc[mi][ni] = wmma_bf16(af[mi], bfv[ni], acc[mi][ni]);

        wait_async0();              // next tile resident in LDS
        __syncthreads();            // everyone done reading `buf`
        buf ^= 1;
    }

    int colb = lane & 15, hi = lane >> 4;
#pragma unroll
    for (int mi = 0; mi < 2; ++mi)
#pragma unroll
        for (int ni = 0; ni < 4; ++ni) {
            int col = bn + wn + ni * 16 + colb;
            float bv = bias ? bias[col] : 0.0f;
#pragma unroll
            for (int j = 0; j < 8; ++j) {
                int row = bm + wm + mi * 16 + j + hi * 8;
                C[(size_t)row * N + col] = acc[mi][ni][j] + bv;
            }
        }
}

// ---------------------------------------------------------------- RoPE + head split
// Qh[h][t][d], Kh[h][t][d] row-major; V written TRANSPOSED: Vt[h][d][t]
__global__ void k_rope_split(const float* __restrict__ qkv, bf16* __restrict__ Qh,
                             bf16* __restrict__ Kh, bf16* __restrict__ Vt) {
    int idx = blockIdx.x * blockDim.x + threadIdx.x;
    if (idx >= SEQ * QKV_N) return;
    int t = idx / QKV_N, c = idx % QKV_N;
    const float LOG_TH_OVER = 9.210340372f / 64.0f;   // ln(10000)/(dim/2)

    if (c < D_MODEL) {                 // Q
        int h = c >> 7, d = c & 127;
        const float* base = qkv + (size_t)t * QKV_N + (h << 7);
        float inv = __expf(-(float)(d & 63) * LOG_TH_OVER);
        float ang = (float)t * inv, s, co;
        __sincosf(ang, &s, &co);
        float out = (d < 64) ? base[d] * co - base[d + 64] * s
                             : base[d] * co + base[d - 64] * s;
        Qh[((size_t)h * SEQ + t) * HEAD_DIM + d] = (bf16)out;
    } else if (c < D_MODEL + 512) {    // K
        int cc = c - D_MODEL;
        int h = cc >> 7, d = cc & 127;
        const float* base = qkv + (size_t)t * QKV_N + D_MODEL + (h << 7);
        float inv = __expf(-(float)(d & 63) * LOG_TH_OVER);
        float ang = (float)t * inv, s, co;
        __sincosf(ang, &s, &co);
        float out = (d < 64) ? base[d] * co - base[d + 64] * s
                             : base[d] * co + base[d - 64] * s;
        Kh[((size_t)h * SEQ + t) * HEAD_DIM + d] = (bf16)out;
    } else {                           // V: store transposed [h][d][t]
        int cc = c - D_MODEL - 512;
        int h = cc >> 7, d = cc & 127;
        Vt[((size_t)h * HEAD_DIM + d) * SEQ + t] = (bf16)qkv[idx];
    }
}

// ---------------------------------------------------------------- flash attention
// 128 threads = 4 waves; Q tile 64 rows (16 rows/wave), K/V tile 64 keys.
#define FA_QT 64
#define FA_KT 64
#define LQ 136   // Q/K row stride (128+8) -> 272B rows, 16B aligned
#define LV 72    // Vs/Ps row stride (64+8) -> 144B rows, 16B aligned

__global__ __launch_bounds__(128)
void k_flash_attn(const bf16* __restrict__ Qh, const bf16* __restrict__ Kh,
                  const bf16* __restrict__ Vt, bf16* __restrict__ ctx) {
    __shared__ bf16 Qs[FA_QT * LQ];
    __shared__ bf16 Ks[FA_KT * LQ];
    __shared__ bf16 Vs[HEAD_DIM * LV];
    __shared__ bf16 Ps[4 * 16 * LV];

    int h   = blockIdx.y;
    int qb  = blockIdx.x * FA_QT;
    int kvh = h >> 2;                          // 4 Q-heads per KV-head
    const bf16* Qp = Qh + (size_t)h   * SEQ * HEAD_DIM;
    const bf16* Kp = Kh + (size_t)kvh * SEQ * HEAD_DIM;
    const bf16* Vp = Vt + (size_t)kvh * HEAD_DIM * SEQ;   // [d][t]

    int tid = threadIdx.x, wid = tid >> 5, lane = tid & 31;
    int colb = lane & 15, hi = lane >> 4;
    const float scale = 0.08838834764831845f;  // 1/sqrt(128)

    // Q tile 64x128, async DMA
#pragma unroll
    for (int i = tid; i < FA_QT * 16; i += 128) {
        int r = i >> 4, c = (i & 15) << 3;
        async_copy_b128(&Qs[r * LQ + c], &Qp[(size_t)(qb + r) * HEAD_DIM + c]);
    }

    float m_i[8], l_i[8];
    f32x8 o[8];
#pragma unroll
    for (int j = 0; j < 8; ++j) { m_i[j] = -1e30f; l_i[j] = 0.0f; }
#pragma unroll
    for (int dt = 0; dt < 8; ++dt) o[dt] = zero8();

    wait_async0();
    __syncthreads();

    int kend = qb + FA_QT;                     // causal bound (exclusive)
    for (int kb = 0; kb < kend; kb += FA_KT) {
        // K tile 64x128 (Bs[n=key][k=d]) and V tile 128x64 (Bs[n=d][k=key]),
        // both contiguous in memory -> pure async DMA
#pragma unroll
        for (int i = tid; i < FA_KT * 16; i += 128) {
            int r = i >> 4, c = (i & 15) << 3;
            async_copy_b128(&Ks[r * LQ + c], &Kp[(size_t)(kb + r) * HEAD_DIM + c]);
        }
#pragma unroll
        for (int i = tid; i < HEAD_DIM * 8; i += 128) {
            int r = i >> 3, c = (i & 7) << 3;
            async_copy_b128(&Vs[r * LV + c], &Vp[(size_t)r * SEQ + kb + c]);
        }
        wait_async0();
        __syncthreads();

        // S = Q(16x128) x K^T(128x64) for this wave's 16 rows
        f32x8 s[4];
#pragma unroll
        for (int nt = 0; nt < 4; ++nt) s[nt] = zero8();
#pragma unroll
        for (int ds = 0; ds < 4; ++ds) {
            bf16x16 a = ldsA_frag(&Qs[(wid * 16) * LQ + ds * 32], LQ, lane);
#pragma unroll
            for (int nt = 0; nt < 4; ++nt) {
                bf16x16 b = ldsB_frag(&Ks[(nt * 16) * LQ + ds * 32], LQ, lane);
                s[nt] = wmma_bf16(a, b, s[nt]);
            }
        }

        // scale + causal mask + row max (8 rows per lane, replicated in 16-lane halves)
        float p[4][8], rmax[8];
#pragma unroll
        for (int j = 0; j < 8; ++j) rmax[j] = -1e30f;
#pragma unroll
        for (int nt = 0; nt < 4; ++nt) {
            int col = kb + nt * 16 + colb;
#pragma unroll
            for (int j = 0; j < 8; ++j) {
                int row = qb + wid * 16 + j + hi * 8;
                float v = s[nt][j] * scale;
                if (col > row) v = -1e30f;
                p[nt][j] = v;
                rmax[j] = fmaxf(rmax[j], v);
            }
        }
#pragma unroll
        for (int msk = 1; msk <= 8; msk <<= 1)
#pragma unroll
            for (int j = 0; j < 8; ++j)
                rmax[j] = fmaxf(rmax[j], __shfl_xor(rmax[j], msk, 32));

        float alpha[8], rsum[8];
#pragma unroll
        for (int j = 0; j < 8; ++j) {
            float mn = fmaxf(m_i[j], rmax[j]);
            alpha[j] = __expf(m_i[j] - mn);
            m_i[j] = mn;
            rsum[j] = 0.0f;
        }
#pragma unroll
        for (int nt = 0; nt < 4; ++nt)
#pragma unroll
            for (int j = 0; j < 8; ++j) {
                float e = __expf(p[nt][j] - m_i[j]);
                p[nt][j] = e;
                rsum[j] += e;
            }
#pragma unroll
        for (int msk = 1; msk <= 8; msk <<= 1)
#pragma unroll
            for (int j = 0; j < 8; ++j)
                rsum[j] += __shfl_xor(rsum[j], msk, 32);
#pragma unroll
        for (int j = 0; j < 8; ++j) l_i[j] = l_i[j] * alpha[j] + rsum[j];
#pragma unroll
        for (int dt = 0; dt < 8; ++dt)
#pragma unroll
            for (int j = 0; j < 8; ++j) o[dt][j] *= alpha[j];

        // stage P (bf16) in per-wave LDS, then O += P(16x64) x V(64x128)
        bf16* myP = &Ps[wid * 16 * LV];
#pragma unroll
        for (int nt = 0; nt < 4; ++nt)
#pragma unroll
            for (int j = 0; j < 8; ++j)
                myP[(j + hi * 8) * LV + nt * 16 + colb] = (bf16)p[nt][j];
        // same-wave LDS RAW: DS ops are in-order within a wave

#pragma unroll
        for (int ks = 0; ks < 2; ++ks) {
            bf16x16 a = ldsA_frag(&myP[ks * 32], LV, lane);
#pragma unroll
            for (int dt = 0; dt < 8; ++dt) {
                bf16x16 b = ldsB_frag(&Vs[(dt * 16) * LV + ks * 32], LV, lane);
                o[dt] = wmma_bf16(a, b, o[dt]);
            }
        }
        __syncthreads();   // before K/V tiles are overwritten
    }

    // epilogue: ctx[t][h*128 + d] = O / l
#pragma unroll
    for (int dt = 0; dt < 8; ++dt)
#pragma unroll
        for (int j = 0; j < 8; ++j) {
            int row = qb + wid * 16 + j + hi * 8;
            int col = dt * 16 + colb;
            ctx[(size_t)row * D_MODEL + h * HEAD_DIM + col] = (bf16)(o[dt][j] / l_i[j]);
        }
}

// ---------------------------------------------------------------- launch
extern "C" void kernel_launch(void* const* d_in, const int* in_sizes, int n_in,
                              void* d_out, int out_size, void* d_ws, size_t ws_size,
                              hipStream_t stream) {
    const float* x  = (const float*)d_in[0];
    const float* Wq = (const float*)d_in[1];
    const float* Wk = (const float*)d_in[2];
    const float* Wv = (const float*)d_in[3];
    const float* Wo = (const float*)d_in[4];
    const float* bo = (const float*)d_in[5];
    float* out = (float*)d_out;

    char* ws = (char*)d_ws;
    size_t off = 0;
    bf16* xb    = (bf16*)(ws + off); off += (size_t)SEQ * D_MODEL * 2;          // 16 MB
    bf16* WcatT = (bf16*)(ws + off); off += (size_t)QKV_N * D_MODEL * 2;        // 12 MB (N-major)
    bf16* WoT   = (bf16*)(ws + off); off += (size_t)D_MODEL * D_MODEL * 2;      //  8 MB (N-major)
    float* qkv  = (float*)(ws + off); off += (size_t)SEQ * QKV_N * 4;           // 48 MB
    bf16* Qh    = (bf16*)(ws + off); off += (size_t)NUM_HEADS * SEQ * HEAD_DIM * 2; // 16 MB
    bf16* Kh    = (bf16*)(ws + off); off += (size_t)KV_HEADS * SEQ * HEAD_DIM * 2;  //  4 MB
    bf16* Vt    = (bf16*)(ws + off); off += (size_t)KV_HEADS * HEAD_DIM * SEQ * 2;  //  4 MB
    bf16* ctx   = xb;  // xb dead after GEMM1; alias

    // 1) casts; weights transposed to N-major so GEMM tiles are contiguous
    {
        int n = SEQ * D_MODEL;
        k_cast_bf16<<<(n + 255) / 256, 256, 0, stream>>>(x, xb, n);
        // WcatT rows: [0,2048) = Wq^T, [2048,2560) = Wk^T, [2560,3072) = Wv^T
        k_cast_transpose<<<dim3(D_MODEL / 32, D_MODEL / 32), 256, 0, stream>>>(
            Wq, WcatT, D_MODEL, D_MODEL);
        k_cast_transpose<<<dim3(KV_OUT / 32, D_MODEL / 32), 256, 0, stream>>>(
            Wk, WcatT + (size_t)D_MODEL * D_MODEL, D_MODEL, KV_OUT);
        k_cast_transpose<<<dim3(KV_OUT / 32, D_MODEL / 32), 256, 0, stream>>>(
            Wv, WcatT + (size_t)(D_MODEL + KV_OUT) * D_MODEL, D_MODEL, KV_OUT);
        k_cast_transpose<<<dim3(D_MODEL / 32, D_MODEL / 32), 256, 0, stream>>>(
            Wo, WoT, D_MODEL, D_MODEL);
    }
    // 2) QKV projection: [4096,2048] x [2048,3072] -> f32
    k_gemm_bf16<<<dim3(QKV_N / GT_N, SEQ / GT_M), 256, 0, stream>>>(
        xb, WcatT, qkv, nullptr, SEQ, QKV_N, D_MODEL);
    // 3) RoPE + split into per-head bf16 layouts (V transposed)
    {
        int n = SEQ * QKV_N;
        k_rope_split<<<(n + 255) / 256, 256, 0, stream>>>(qkv, Qh, Kh, Vt);
    }
    // 4) causal flash attention per (q-tile, head)
    k_flash_attn<<<dim3(SEQ / FA_QT, NUM_HEADS), 128, 0, stream>>>(Qh, Kh, Vt, ctx);
    // 5) output projection with bias: [4096,2048] x [2048,2048] + bo -> d_out
    k_gemm_bf16<<<dim3(D_MODEL / GT_N, SEQ / GT_M), 256, 0, stream>>>(
        ctx, WoT, out, bo, SEQ, D_MODEL, D_MODEL);
}
